// CLAM_SB_40750649705194
// MI455X (gfx1250) — compile-verified, matrix operand order checked by reading, not run on hardware
//
#include <hip/hip_runtime.h>
#include <hip/hip_bf16.h>

#define FEAT 1024
#define HID 256
#define KSAMP 8
#define TAU 1.0f
#define ALPHA 1.0f

typedef __attribute__((ext_vector_type(16))) __bf16 v16bf;
typedef __attribute__((ext_vector_type(8)))  __bf16 v8bf;
typedef __attribute__((ext_vector_type(4)))  __bf16 v4bf;
typedef __attribute__((ext_vector_type(8)))  float  v8f;

// 16-element bf16 fragment (CDNA5 16-bit A/B layout slice) converted on the
// fly from fp32 memory: 8 consecutive floats at p, 8 more at p+16.
__device__ __forceinline__ v16bf cvt_frag_f32(const float* __restrict__ p) {
  const float4* q = (const float4*)p;
  float4 f0 = q[0], f1 = q[1], f2 = q[4], f3 = q[5];
  v16bf v;
  v[0]  = (__bf16)f0.x; v[1]  = (__bf16)f0.y; v[2]  = (__bf16)f0.z; v[3]  = (__bf16)f0.w;
  v[4]  = (__bf16)f1.x; v[5]  = (__bf16)f1.y; v[6]  = (__bf16)f1.z; v[7]  = (__bf16)f1.w;
  v[8]  = (__bf16)f2.x; v[9]  = (__bf16)f2.y; v[10] = (__bf16)f2.z; v[11] = (__bf16)f2.w;
  v[12] = (__bf16)f3.x; v[13] = (__bf16)f3.y; v[14] = (__bf16)f3.z; v[15] = (__bf16)f3.w;
  return v;
}

// Same fragment shape from bf16 memory (global workspace or LDS): two b128s.
__device__ __forceinline__ v16bf frag_bf16(const __bf16* p) {
  v8bf lo = *(const v8bf*)p;
  v8bf hi = *(const v8bf*)(p + 16);
  v16bf v;
#pragma unroll
  for (int i = 0; i < 8; ++i) { v[i] = lo[i]; v[8 + i] = hi[i]; }
  return v;
}

__device__ __forceinline__ v8f wmma_bf16(v16bf a, v16bf b, v8f c) {
  return __builtin_amdgcn_wmma_f32_16x16x32_bf16(false, a, false, b, (short)0, c,
                                                 false, false);
}

// ---------------------------------------------------------------------------
// Kernel 0: one-shot fp32 -> bf16 conversion of W_fc / Wa / Wb into workspace
// (768 KB total -> stays resident in the 192 MB L2 for all GEMM WGs).
// ---------------------------------------------------------------------------
__global__ __launch_bounds__(256) void clam_prep_weights(
    const float* __restrict__ Wfc, const float* __restrict__ Wa,
    const float* __restrict__ Wb, __bf16* __restrict__ wfc_b,
    __bf16* __restrict__ wa_b, __bf16* __restrict__ wb_b)
{
  const int i = blockIdx.x * 256 + threadIdx.x;   // index of 4-element group
  const int NW1 = HID * FEAT / 4;                 // 65536 groups
  const int NW2 = HID * HID / 4;                  // 16384 groups
  const float* src;
  __bf16* dst;
  int g;
  if (i < NW1)            { src = Wfc; dst = wfc_b; g = i; }
  else if (i < NW1 + NW2) { src = Wa;  dst = wa_b;  g = i - NW1; }
  else                    { src = Wb;  dst = wb_b;  g = i - NW1 - NW2; }
  const float4 f = ((const float4*)src)[g];
  v4bf o;
  o[0] = (__bf16)f.x; o[1] = (__bf16)f.y; o[2] = (__bf16)f.z; o[3] = (__bf16)f.w;
  *(v4bf*)(dst + (size_t)g * 4) = o;
}

// ---------------------------------------------------------------------------
// Kernel 1: fused  x = relu(h@W_fc^T + b)  ->  gated attention score per row.
// 256 threads = 8 waves; WG tile = 64 rows x 256 cols.
// wave w: m-tile = w&3 (16 rows), n-half = w>>2 (128 cols = 8 WMMA n-tiles).
// B fragments are batch-loaded ahead of the WMMA burst so the waits stagger
// (s_wait_loadcnt N..0) instead of a full drain before every WMMA.
// ---------------------------------------------------------------------------
__global__ __launch_bounds__(256) void clam_fused_x_attn(
    const float* __restrict__ h, const __bf16* __restrict__ Wfc_b,
    const float* __restrict__ b_fc, const __bf16* __restrict__ Wa_b,
    const float* __restrict__ ba, const __bf16* __restrict__ Wb_b,
    const float* __restrict__ bb, const float* __restrict__ Wc,
    const float* __restrict__ bc, __bf16* __restrict__ xbuf,
    float* __restrict__ scores, int Ntot)
{
  __shared__ __bf16 xs[64 * HID];   // 32 KB x tile (bf16)
  __shared__ float  sc[64];         // per-row attention score partials

  const int tid  = threadIdx.x;
  const int lane = tid & 31;
  const int wv   = tid >> 5;
  const int mt   = wv & 3;          // which 16-row tile
  const int nh   = wv >> 2;         // which 128-col half
  const int rsel = lane & 15;       // row / col within 16-tile
  const int khi  = lane >> 4;       // K-half select (A/B 16-bit layout)

  if (tid < 64) sc[tid] = 0.f;

  const long row0 = (long)blockIdx.x * 64;
  long ar = row0 + mt * 16 + rsel;
  if (ar > Ntot - 1) ar = Ntot - 1;              // clamp tail rows (stores guarded)
  const float* hrow = h + (size_t)ar * FEAT + khi * 8;
  const __bf16* wfc0 = Wfc_b + (size_t)(nh * 128 + rsel) * FEAT + khi * 8;

  // ---- stage 1: x = relu(h @ W_fc^T + b_fc), K = 1024 ----
  v8f acc[8] = {};
  for (int k0 = 0; k0 < FEAT; k0 += 32) {
    __builtin_prefetch(hrow + k0 + 128, 0, 0);   // speculative; OOB dropped
    v16bf af = cvt_frag_f32(hrow + k0);
    v16bf bw[8];
#pragma unroll
    for (int t = 0; t < 8; ++t)
      bw[t] = frag_bf16(wfc0 + (size_t)t * 16 * FEAT + k0);
#pragma unroll
    for (int t = 0; t < 8; ++t)
      acc[t] = wmma_bf16(af, bw[t], acc[t]);
  }
  // bias + relu, scatter D-layout (lane<16: M=v, lane>=16: M=v+8) into LDS
#pragma unroll
  for (int t = 0; t < 8; ++t) {
    const int n = nh * 128 + t * 16 + rsel;
    const float bn = b_fc[n];
#pragma unroll
    for (int v = 0; v < 8; ++v) {
      const int m = mt * 16 + khi * 8 + v;
      float xv = acc[t][v] + bn;
      xs[m * HID + n] = (__bf16)(xv > 0.f ? xv : 0.f);
    }
  }
  __syncthreads();

  // coalesced bf16 spill of the x tile to global workspace (b128 stores)
  for (int g = tid; g < 64 * HID / 8; g += 256) {
    const int m = g >> 5;
    const int c = (g & 31) * 8;
    const long gr = row0 + m;
    if (gr < Ntot)
      *(v8bf*)(xbuf + (size_t)gr * HID + c) = *(const v8bf*)(xs + m * HID + c);
  }

  // ---- stage 2: a = tanh(x@Wa^T), g = sigmoid(x@Wb^T), s = (a*g)·Wc ----
  v8f aacc[8] = {}, gacc[8] = {};
  const __bf16* xrow = xs + (mt * 16 + rsel) * HID + khi * 8;
  const __bf16* wa0 = Wa_b + (size_t)(nh * 128 + rsel) * HID + khi * 8;
  const __bf16* wb0 = Wb_b + (size_t)(nh * 128 + rsel) * HID + khi * 8;
  for (int k0 = 0; k0 < HID; k0 += 32) {
    v16bf af = frag_bf16(xrow + k0);
    // two half-batches of 8 fragments each keeps peak VGPRs ~200 (no spill)
#pragma unroll
    for (int half = 0; half < 2; ++half) {
      v16bf wf[8];
#pragma unroll
      for (int t = 0; t < 4; ++t) {
        const size_t roff = (size_t)(half * 4 + t) * 16 * HID + k0;
        wf[2 * t]     = frag_bf16(wa0 + roff);
        wf[2 * t + 1] = frag_bf16(wb0 + roff);
      }
#pragma unroll
      for (int t = 0; t < 4; ++t) {
        aacc[half * 4 + t] = wmma_bf16(af, wf[2 * t],     aacc[half * 4 + t]);
        gacc[half * 4 + t] = wmma_bf16(af, wf[2 * t + 1], gacc[half * 4 + t]);
      }
    }
  }
  float part[8] = {};
#pragma unroll
  for (int t = 0; t < 8; ++t) {
    const int n = nh * 128 + t * 16 + rsel;
    const float ban = ba[n], bbn = bb[n], wcn = Wc[n];
#pragma unroll
    for (int v = 0; v < 8; ++v) {
      const float av = tanhf(aacc[t][v] + ban);
      const float gv = 1.f / (1.f + expf(-(gacc[t][v] + bbn)));
      part[v] += av * gv * wcn;
    }
  }
#pragma unroll
  for (int v = 0; v < 8; ++v)
    atomicAdd(&sc[mt * 16 + khi * 8 + v], part[v]);
  __syncthreads();
  if (tid < 64) {
    const long r = row0 + tid;
    if (r < Ntot) scores[r] = sc[tid] + bc[0];
  }
}

// ---------------------------------------------------------------------------
// Kernel 2: single WG — softmax stats + top-8 / bottom-8 selection.
// ---------------------------------------------------------------------------
__global__ __launch_bounds__(1024) void clam_reduce_topk(
    const float* __restrict__ scores, float* __restrict__ topc,
    float* __restrict__ botc, float* __restrict__ stats,
    float* __restrict__ Mbuf, int* __restrict__ ids, int Ntot)
{
  __shared__ float sval[1024];
  __shared__ int   sidx[1024];
  const int t = threadIdx.x;

  float mx = -3.4e38f;
  for (int i = t; i < Ntot; i += 1024) {
    const float s = scores[i];
    topc[i] = s; botc[i] = s;
    mx = fmaxf(mx, s);
  }
  sval[t] = mx; __syncthreads();
  for (int s = 512; s > 0; s >>= 1) { if (t < s) sval[t] = fmaxf(sval[t], sval[t + s]); __syncthreads(); }
  const float gmax = sval[0]; __syncthreads();

  float sum = 0.f;
  for (int i = t; i < Ntot; i += 1024) sum += expf(scores[i] - gmax);
  sval[t] = sum; __syncthreads();
  for (int s = 512; s > 0; s >>= 1) { if (t < s) sval[t] += sval[t + s]; __syncthreads(); }
  if (t == 0) { stats[0] = gmax; stats[1] = sval[0]; }
  if (t < HID) Mbuf[t] = 0.f;           // zero pooling accumulator for kernel 3
  __syncthreads();

  for (int sel = 0; sel < KSAMP; ++sel) {        // top-8 (argmax, low index wins ties)
    float bv = -3.4e38f; int bi = 0x7fffffff;
    for (int i = t; i < Ntot; i += 1024) { const float v = topc[i]; if (v > bv) { bv = v; bi = i; } }
    sval[t] = bv; sidx[t] = bi; __syncthreads();
    for (int s = 512; s > 0; s >>= 1) {
      if (t < s && (sval[t + s] > sval[t] || (sval[t + s] == sval[t] && sidx[t + s] < sidx[t]))) {
        sval[t] = sval[t + s]; sidx[t] = sidx[t + s];
      }
      __syncthreads();
    }
    if (t == 0) { ids[sel] = sidx[0]; topc[sidx[0]] = -3.4e38f; }
    __syncthreads();
  }
  for (int sel = 0; sel < KSAMP; ++sel) {        // bottom-8 (argmin)
    float bv = 3.4e38f; int bi = 0x7fffffff;
    for (int i = t; i < Ntot; i += 1024) { const float v = botc[i]; if (v < bv) { bv = v; bi = i; } }
    sval[t] = bv; sidx[t] = bi; __syncthreads();
    for (int s = 512; s > 0; s >>= 1) {
      if (t < s && (sval[t + s] < sval[t] || (sval[t + s] == sval[t] && sidx[t + s] < sidx[t]))) {
        sval[t] = sval[t + s]; sidx[t] = sidx[t + s];
      }
      __syncthreads();
    }
    if (t == 0) { ids[KSAMP + sel] = sidx[0]; botc[sidx[0]] = 3.4e38f; }
    __syncthreads();
  }
}

// ---------------------------------------------------------------------------
// Kernel 3: attention pooling M = sum_i softmax(s)_i * x_i   (streams x bf16)
// ---------------------------------------------------------------------------
__global__ __launch_bounds__(256) void clam_pool(
    const __bf16* __restrict__ xbuf, const float* __restrict__ scores,
    const float* __restrict__ stats, float* __restrict__ Mbuf, int Ntot)
{
  __shared__ float lm[HID];
  const int t  = threadIdx.x;
  const int cg = t & 31, rs = t >> 5;
  const float gmax = stats[0], inv = 1.f / stats[1];
  float acc[8] = {};
  for (long r = (long)blockIdx.x * 8 + rs; r < Ntot; r += (long)gridDim.x * 8) {
    const float wgt = expf(scores[r] - gmax) * inv;
    v8bf xv = *(const v8bf*)(xbuf + (size_t)r * HID + cg * 8);
#pragma unroll
    for (int i = 0; i < 8; ++i) acc[i] += wgt * (float)xv[i];
  }
  lm[t] = 0.f; __syncthreads();
#pragma unroll
  for (int i = 0; i < 8; ++i) atomicAdd(&lm[cg * 8 + i], acc[i]);
  __syncthreads();
  atomicAdd(&Mbuf[t], lm[t]);
}

// ---------------------------------------------------------------------------
// Kernel 4: bag classifier + smooth-top1-SVM instance loss; writes 6 outputs:
// logits[2], probs[2], argmax, inst_loss.
// ---------------------------------------------------------------------------
__global__ __launch_bounds__(256) void clam_finalize(
    const __bf16* __restrict__ xbuf, const float* __restrict__ Mbuf,
    const int* __restrict__ ids, const float* __restrict__ W_cls,
    const float* __restrict__ b_cls, const float* __restrict__ W_inst,
    const float* __restrict__ b_inst, const int* __restrict__ label_p,
    float* __restrict__ out)
{
  __shared__ float l2[2];
  __shared__ float loss;
  const int t = threadIdx.x;
  if (t < 2) l2[t] = b_cls[t];
  if (t == 0) loss = 0.f;
  __syncthreads();

  const float m = Mbuf[t];
  atomicAdd(&l2[0], m * W_cls[t]);
  atomicAdd(&l2[1], m * W_cls[HID + t]);

  const int label = label_p[0];
  if (t < 2 * KSAMP) {
    const int id = ids[t];
    const __bf16* xi = xbuf + (size_t)id * HID;
    const float* w0 = W_inst + (size_t)label * 2 * HID;
    const float* w1 = w0 + HID;
    float s0 = b_inst[label * 2 + 0], s1 = b_inst[label * 2 + 1];
    for (int k = 0; k < HID; ++k) {
      const float xv = (float)xi[k];
      s0 += xv * w0[k]; s1 += xv * w1[k];
    }
    const int tgt = (t < KSAMP) ? 1 : 0;     // top-k -> class 1, bottom-k -> 0
    const float d0 = (tgt == 0) ? 0.f : ALPHA;
    const float d1 = (tgt == 1) ? 0.f : ALPHA;
    const float z0 = (s0 + d0) / TAU, z1 = (s1 + d1) / TAU;
    const float zm = fmaxf(z0, z1);
    const float lse = TAU * (zm + logf(expf(z0 - zm) + expf(z1 - zm)));
    const float sy = tgt ? s1 : s0;
    atomicAdd(&loss, lse - sy);
  }
  __syncthreads();
  if (t == 0) {
    const float l0 = l2[0], l1 = l2[1];
    out[0] = l0; out[1] = l1;
    const float zm = fmaxf(l0, l1);
    const float e0 = expf(l0 - zm), e1 = expf(l1 - zm);
    const float s = e0 + e1;
    out[2] = e0 / s; out[3] = e1 / s;
    out[4] = (l1 > l0) ? 1.f : 0.f;
    out[5] = loss / (float)(2 * KSAMP);
  }
}

extern "C" void kernel_launch(void* const* d_in, const int* in_sizes, int n_in,
                              void* d_out, int out_size, void* d_ws, size_t ws_size,
                              hipStream_t stream)
{
  const float* h      = (const float*)d_in[0];
  const int*   label  = (const int*)d_in[1];
  const float* W_fc   = (const float*)d_in[2];
  const float* b_fc   = (const float*)d_in[3];
  const float* Wa     = (const float*)d_in[4];
  const float* ba     = (const float*)d_in[5];
  const float* Wb     = (const float*)d_in[6];
  const float* bb     = (const float*)d_in[7];
  const float* Wc     = (const float*)d_in[8];
  const float* bc     = (const float*)d_in[9];
  const float* W_cls  = (const float*)d_in[10];
  const float* b_cls  = (const float*)d_in[11];
  const float* W_inst = (const float*)d_in[12];
  const float* b_inst = (const float*)d_in[13];
  const int N = in_sizes[0] / FEAT;

  char* w = (char*)d_ws;
  size_t off = 0;
  __bf16* xbuf  = (__bf16*)(w + off); off += (size_t)N * HID * 2;    off = (off + 255) & ~(size_t)255;
  __bf16* wfc_b = (__bf16*)(w + off); off += (size_t)HID * FEAT * 2; off = (off + 255) & ~(size_t)255;
  __bf16* wa_b  = (__bf16*)(w + off); off += (size_t)HID * HID * 2;  off = (off + 255) & ~(size_t)255;
  __bf16* wb_b  = (__bf16*)(w + off); off += (size_t)HID * HID * 2;  off = (off + 255) & ~(size_t)255;
  float* scores = (float*)(w + off);  off += (size_t)N * 4;          off = (off + 255) & ~(size_t)255;
  float* topc   = (float*)(w + off);  off += (size_t)N * 4;          off = (off + 255) & ~(size_t)255;
  float* botc   = (float*)(w + off);  off += (size_t)N * 4;          off = (off + 255) & ~(size_t)255;
  float* stats  = (float*)(w + off);  off += 2 * 4;                  off = (off + 255) & ~(size_t)255;
  float* Mbuf   = (float*)(w + off);  off += HID * 4;                off = (off + 255) & ~(size_t)255;
  int*   ids    = (int*)(w + off);    off += 2 * KSAMP * 4;
  (void)n_in; (void)out_size; (void)ws_size;

  const int prep_groups = (HID * FEAT + 2 * HID * HID) / 4;   // 98304
  clam_prep_weights<<<prep_groups / 256, 256, 0, stream>>>(W_fc, Wa, Wb,
                                                           wfc_b, wa_b, wb_b);
  const int blocks1 = (N + 63) / 64;
  clam_fused_x_attn<<<blocks1, 256, 0, stream>>>(h, wfc_b, b_fc, wa_b, ba, wb_b,
                                                 bb, Wc, bc, xbuf, scores, N);
  clam_reduce_topk<<<1, 1024, 0, stream>>>(scores, topc, botc, stats, Mbuf, ids, N);
  clam_pool<<<512, 256, 0, stream>>>(xbuf, scores, stats, Mbuf, N);
  clam_finalize<<<1, 256, 0, stream>>>(xbuf, Mbuf, ids, W_cls, b_cls, W_inst,
                                       b_inst, label, (float*)d_out);
}
